// TextCNNVQ3_22582938042459
// MI455X (gfx1250) — matrix operand order repaired and checked by reading, not compiled, not run on hardware
//
#include <hip/hip_runtime.h>
#include <hip/hip_bf16.h>
#include <stdint.h>
#include <utility>

// ---------------------------------------------------------------------------
// Model constants (from the reference): V,E,CH,RC,VQ,K,B,L = 256,256,512,256,8,512,16,1024
#define B_SZ  16
#define L_SZ  1024
#define V_SZ  256
#define E_SZ  256
#define CH_SZ 512
#define RC_SZ 256
#define VQ_SZ 8
#define KCB   512

typedef __attribute__((ext_vector_type(16))) __bf16 v16bf;
typedef __attribute__((ext_vector_type(8)))  float  v8f;

// ---------------------------------------------------------------------------
// small device helpers
__device__ __forceinline__ uint16_t f2bf(float f) {
    union { float f; uint32_t u; } v; v.f = f;
    uint32_t u = v.u;
    uint32_t r = u + 0x7FFFu + ((u >> 16) & 1u);   // round-to-nearest-even
    return (uint16_t)(r >> 16);
}
__device__ __forceinline__ float bf2f(uint16_t h) {
    union { uint32_t u; float f; } v; v.u = ((uint32_t)h) << 16; return v.f;
}
__device__ __forceinline__ float eluf(float x) { return x > 0.f ? x : (__expf(x) - 1.f); }

// ---------------------------------------------------------------------------
// Generic implicit-GEMM conv kernel on the bf16 WMMA path.
//   Y[b, co, n*outStride+outOffset] = bias[co] + sum_{ci,tap} W[co, ci*kw+tap] * X[b, ci, n*stride+tap-pad]
// Workgroup tile: M=64 x N=64.  4 waves; wave w owns N-strip w (16 cols) and
// walks 4 M-subtiles with 4 independent f32 accumulators -> 16 v_wmma per
// K-chunk per workgroup, B-fragment reused across the 4 WMMAs of a wave.
// kw and eluIn are template params so the im2col addressing is shift/mul only.
struct ConvArgs {
    const uint16_t* X;    // bf16 [B, Cin, Tin]
    const uint16_t* Wp;   // bf16 [Cout, Ktot]   (Ktot = Cin*kw, tap-minor)
    const float*    bias; // [Cout]
    float*          Yf;   // f32 [B, Cout, ToutFull] (optional)
    uint16_t*       Yh;   // bf16 same (optional)
    int Cin, Cout, Tin, ToutFull, Nout;
    int stride, pad;
    int outStride, outOffset;
    int eluOut;
    int Ktot;
};

template <int KW, bool ELUIN>
__global__ __launch_bounds__(128) void conv_wmma_kernel(ConvArgs a) {
    __shared__ uint32_t Atile[64 * 16];  // 64 Cout rows x 32 K (bf16 pairs)
    __shared__ uint32_t Btile[64 * 16];  // 64 cols     x 32 K

    const int tid  = threadIdx.x;
    const int lane = tid & 31;
    const int wave = tid >> 5;
    const int b      = blockIdx.z;
    const int coBase = blockIdx.y * 64;
    const int nBase  = blockIdx.x * 64;
    const int half   = lane >> 4;
    const int l15    = lane & 15;

    v8f acc[4] = {};
    const int Kt = a.Ktot;
    const int nChunks = (Kt + 31) >> 5;

    for (int kb = 0; kb < nChunks; ++kb) {
        __syncthreads();
        // ---- stage A (weights): 64x32 bf16 = 1024 dwords
        for (int t = tid; t < 64 * 16; t += 128) {
            int row = t >> 4, ui = t & 15;
            int K0 = kb * 32 + ui * 2;
            int co = coBase + row;
            uint32_t pv = 0;
            if (co < a.Cout) {
                uint16_t e0 = (K0     < Kt) ? a.Wp[(size_t)co * Kt + K0]     : (uint16_t)0;
                uint16_t e1 = (K0 + 1 < Kt) ? a.Wp[(size_t)co * Kt + K0 + 1] : (uint16_t)0;
                pv = ((uint32_t)e1 << 16) | e0;
            }
            Atile[t] = pv;
        }
        // hint: pull next weight chunk toward the caches (global_prefetch_b8)
        if (kb + 1 < nChunks) {
            int co = coBase + (tid & 63);
            if (co < a.Cout)
                __builtin_prefetch(&a.Wp[(size_t)co * Kt + (kb + 1) * 32], 0, 1);
        }
        // ---- stage B (im2col of activations): 64 cols x 32 K = 1024 dwords
        for (int t = tid; t < 64 * 16; t += 128) {
            int nl = t >> 4, ui = t & 15;
            int n  = nBase + nl;
            int K0 = kb * 32 + ui * 2;
            uint32_t pv = 0;
            if (n < a.Nout) {
                uint16_t e[2];
#pragma unroll
                for (int q = 0; q < 2; ++q) {
                    int K = K0 + q;
                    uint16_t val = 0;
                    if (K < Kt) {
                        int ci = K / KW, tap = K - ci * KW;   // compile-time KW
                        int pos = n * a.stride + tap - a.pad;
                        if (pos >= 0 && pos < a.Tin) {
                            val = a.X[((size_t)b * a.Cin + ci) * a.Tin + pos];
                            if (ELUIN) val = f2bf(eluf(bf2f(val)));
                        }
                    }
                    e[q] = val;
                }
                pv = ((uint32_t)e[1] << 16) | e[0];
            }
            Btile[t] = pv;
        }
        __syncthreads();

        // ---- fragments per ISA 7.12.2 wave32 layouts
        union frag_t { v16bf v; uint32_t u[8]; };
        frag_t fb;
        const int ncol = wave * 16 + l15;
#pragma unroll
        for (int v = 0; v < 8; ++v)
            fb.u[v] = Btile[ncol * 16 + (v >> 2) * 8 + half * 4 + (v & 3)];   // K = (v>>2)*16 + half*8 + 2(v&3)
#pragma unroll
        for (int m = 0; m < 4; ++m) {
            frag_t fa;
            const int row = m * 16 + l15;
#pragma unroll
            for (int i = 0; i < 4; ++i) {
                fa.u[i]     = Atile[row * 16 + half * 4 + i];       // K = half*8 + 2i
                fa.u[4 + i] = Atile[row * 16 + 8 + half * 4 + i];   // K = 16 + half*8 + 2i
            }
            acc[m] = __builtin_amdgcn_wmma_f32_16x16x32_bf16(
                false, fa.v, false, fb.v, (short)0, acc[m], false, false);
        }
    }

    // ---- store D (16x16 f32 layout: VGPR r -> M = r + 8*half, N = lane&15)
    const int ncol = nBase + wave * 16 + l15;
    if (ncol < a.Nout) {
        const int oIdx = ncol * a.outStride + a.outOffset;
        if (oIdx < a.ToutFull) {
#pragma unroll
            for (int m = 0; m < 4; ++m) {
#pragma unroll
                for (int r = 0; r < 8; ++r) {
                    int co = coBase + m * 16 + r + half * 8;
                    if (co < a.Cout) {
                        float y = acc[m][r] + a.bias[co];
                        if (a.eluOut) y = eluf(y);
                        size_t off = ((size_t)b * a.Cout + co) * a.ToutFull + oIdx;
                        if (a.Yf) a.Yf[off] = y;
                        if (a.Yh) a.Yh[off] = f2bf(y);
                    }
                }
            }
        }
    }
}

// ---------------------------------------------------------------------------
// weight packing
__global__ void cast_f32_bf16_kernel(const float* __restrict__ W, uint16_t* __restrict__ Wp, int n) {
    int i = blockIdx.x * 256 + threadIdx.x;
    if (i < n) Wp[i] = f2bf(W[i]);
}
// ConvTranspose1d(k=4,s=2,p=1), torch layout W[Cin][Cout][4] ->
//   even phase (y[2n] = x[n-1]*w3 + x[n]*w1) and odd phase (y[2n+1] = x[n]*w2 + x[n+1]*w0)
__global__ void pack_convT_kernel(const float* __restrict__ W, uint16_t* __restrict__ Wev,
                                  uint16_t* __restrict__ Wod, int Cin, int Cout) {
    int i = blockIdx.x * 256 + threadIdx.x;
    if (i >= Cin * Cout) return;
    int ci = i / Cout, co = i - ci * Cout;
    const float* w = W + ((size_t)ci * Cout + co) * 4;
    size_t base = ((size_t)co * Cin + ci) * 2;
    Wev[base + 0] = f2bf(w[3]); Wev[base + 1] = f2bf(w[1]);
    Wod[base + 0] = f2bf(w[2]); Wod[base + 1] = f2bf(w[0]);
}

// ---------------------------------------------------------------------------
// embedding gather + max_norm=1 renorm -> h [B,E,L] (f32 + bf16)
__global__ void embed_kernel(const float* __restrict__ emb, const int* __restrict__ x,
                             float* __restrict__ hF, uint16_t* __restrict__ hH,
                             int Bn, int Ln, int En) {
    int i = blockIdx.x * 256 + threadIdx.x;
    if (i >= Bn * Ln) return;
    int b = i / Ln, l = i - b * Ln;
    const float* e = emb + (size_t)x[i] * En;
    float ss = 0.f;
    for (int d = 0; d < En; ++d) ss += e[d] * e[d];
    float s = fminf(1.f, 1.f / (sqrtf(ss) + 1e-7f));
    for (int d = 0; d < En; ++d) {
        float v = e[d] * s;
        size_t off = ((size_t)b * En + d) * Ln + l;
        hF[off] = v; hH[off] = f2bf(v);
    }
}

// channel-wise layernorm of (Y + X) over C, per (b,t) column; in-place safe
__global__ __launch_bounds__(128) void cwln_kernel(const float* __restrict__ Y,
                                                   const float* __restrict__ X,
                                                   const float* __restrict__ g,
                                                   const float* __restrict__ beta,
                                                   float* __restrict__ OutF,
                                                   uint16_t* __restrict__ OutH,
                                                   int C, int T) {
    __shared__ float red[128];
    int bt = blockIdx.x;
    int b = bt / T, t = bt - b * T;
    const size_t base = (size_t)b * C * T + t;
    int tid = threadIdx.x;
    float s = 0.f;
    for (int c = tid; c < C; c += 128) s += Y[base + (size_t)c * T] + X[base + (size_t)c * T];
    red[tid] = s; __syncthreads();
    for (int o = 64; o > 0; o >>= 1) { if (tid < o) red[tid] += red[tid + o]; __syncthreads(); }
    float mu = red[0] / C; __syncthreads();
    float v = 0.f;
    for (int c = tid; c < C; c += 128) {
        float d = Y[base + (size_t)c * T] + X[base + (size_t)c * T] - mu; v += d * d;
    }
    red[tid] = v; __syncthreads();
    for (int o = 64; o > 0; o >>= 1) { if (tid < o) red[tid] += red[tid + o]; __syncthreads(); }
    float rstd = rsqrtf(red[0] / C + 1e-5f);
    for (int c = tid; c < C; c += 128) {
        float d = (Y[base + (size_t)c * T] + X[base + (size_t)c * T] - mu) * rstd * g[c] + beta[c];
        OutF[base + (size_t)c * T] = d;
        OutH[base + (size_t)c * T] = f2bf(d);
    }
}

// VQ: tanh, nearest code over K=512 (D=8), code out + mean-squared diff accumulation
__global__ void vq_kernel(const float* __restrict__ Q, const float* __restrict__ cb,
                          float* __restrict__ codeOut, float* __restrict__ diffAcc,
                          int Bn, int D, int T, int Kc, float scale) {
    int i = blockIdx.x * 256 + threadIdx.x;
    if (i >= Bn * T) return;
    int b = i / T, t = i - b * T;
    float v[VQ_SZ];
    float v2 = 0.f;
    for (int d = 0; d < D; ++d) {
        v[d] = tanhf(Q[((size_t)b * D + d) * T + t]);
        v2 += v[d] * v[d];
    }
    int best = 0; float bestd = 3.4e38f;
    for (int k = 0; k < Kc; ++k) {
        float dot = 0.f, e2 = 0.f;
        for (int d = 0; d < D; ++d) {
            float e = cb[(size_t)d * Kc + k];
            dot += v[d] * e; e2 += e * e;
        }
        float dist = v2 - 2.f * dot + e2;
        if (dist < bestd) { bestd = dist; best = k; }   // first-min, like argmin
    }
    codeOut[i] = (float)best;
    float dd = 0.f;
    for (int d = 0; d < D; ++d) { float df = cb[(size_t)d * Kc + best] - v[d]; dd += df * df; }
    atomicAdd(diffAcc, dd * scale);
}

__global__ void cat_copy_kernel(const uint16_t* __restrict__ src, uint16_t* __restrict__ dst,
                                int Bn, int Csrc, int T, int Cdst, int chOff) {
    size_t i = (size_t)blockIdx.x * 256 + threadIdx.x;
    size_t n = (size_t)Bn * Csrc * T;
    if (i >= n) return;
    int t = (int)(i % T); size_t r = i / T;
    int c = (int)(r % Csrc); int b = (int)(r / Csrc);
    dst[((size_t)b * Cdst + chOff + c) * T + t] = src[i];
}

__global__ void zero_kernel(float* p, int n) {
    int i = blockIdx.x * 256 + threadIdx.x;
    if (i < n) p[i] = 0.f;
}
__global__ void diff_finalize_kernel(const float* acc, float* out) {
    if (threadIdx.x == 0 && blockIdx.x == 0) out[0] = acc[0] + acc[1];
}

// log_softmax over V=256: one wave per (b,t), 8 values per lane, shfl reduction
__global__ __launch_bounds__(128) void logsoftmax_kernel(const float* __restrict__ X,
                                                         float* __restrict__ Out,
                                                         int Bn, int Vn, int T) {
    int wid  = (blockIdx.x * 128 + threadIdx.x) >> 5;
    int lane = threadIdx.x & 31;
    if (wid >= Bn * T) return;
    int b = wid / T, t = wid - b * T;
    const size_t base = (size_t)b * Vn * T + t;
    float vals[8];
    float mx = -3.4e38f;
    for (int j = 0; j < 8; ++j) {
        vals[j] = X[base + (size_t)(j * 32 + lane) * T];
        mx = fmaxf(mx, vals[j]);
    }
    for (int o = 16; o; o >>= 1) mx = fmaxf(mx, __shfl_xor(mx, o, 32));
    float s = 0.f;
    for (int j = 0; j < 8; ++j) s += __expf(vals[j] - mx);
    for (int o = 16; o; o >>= 1) s += __shfl_xor(s, o, 32);
    float lse = mx + __logf(s);
    for (int j = 0; j < 8; ++j)
        Out[((size_t)b * T + t) * Vn + j * 32 + lane] = vals[j] - lse;
}

// ---------------------------------------------------------------------------
extern "C" void kernel_launch(void* const* d_in, const int* in_sizes, int n_in,
                              void* d_out, int out_size, void* d_ws, size_t ws_size,
                              hipStream_t stream) {
    (void)in_sizes; (void)n_in; (void)out_size; (void)ws_size;
    // JAX pytree (sorted dict key) leaf order — see analysis for the full walk.
    auto Fp = [&](int i) -> const float* { return (const float*)d_in[i]; };
    const int* xin = (const int*)d_in[151];
    float* out = (float*)d_out;

    // ---- workspace bump allocator
    uint8_t* ws = (uint8_t*)d_ws;
    size_t off = 0;
    auto alloc = [&](size_t bytes) -> void* {
        void* p = ws + off;
        off = (off + bytes + 255) & ~(size_t)255;
        return p;
    };
    const size_t ACT = (size_t)B_SZ * CH_SZ * L_SZ;   // largest activation
    float* AF[6]; uint16_t* AH[6];
    for (int i = 0; i < 6; ++i) {
        AF[i] = (float*)alloc(ACT * 4);
        AH[i] = (uint16_t*)alloc(ACT * 2);
    }
    uint16_t* WPK  = (uint16_t*)alloc((size_t)CH_SZ * CH_SZ * 4 * 2);  // packed weights
    uint16_t* WPK2 = (uint16_t*)alloc((size_t)CH_SZ * CH_SZ * 2 * 2);  // convT odd phase
    const size_t SMALL = (size_t)B_SZ * VQ_SZ * 256;
    float* QTf = (float*)alloc(SMALL * 4); uint16_t* QTh = (uint16_t*)alloc(SMALL * 2);
    float* QBf = (float*)alloc(SMALL * 4); uint16_t* QBh = (uint16_t*)alloc(SMALL * 2);
    float* CDf = (float*)alloc(SMALL * 4); uint16_t* CDh = (uint16_t*)alloc(SMALL * 2);
    float* X1f = (float*)alloc(SMALL * 4); uint16_t* X1h = (uint16_t*)alloc(SMALL * 2);
    uint16_t* CATh  = (uint16_t*)alloc((size_t)B_SZ * (CH_SZ + VQ_SZ) * 256 * 2);
    uint16_t* CAT2h = (uint16_t*)alloc((size_t)B_SZ * 2 * VQ_SZ * 256 * 2);
    float* diffAcc = (float*)alloc(2 * 4);

    // ---- launch helpers
    auto conv_raw = [&](const uint16_t* X, const uint16_t* Wp, const float* bias,
                        float* Yf, uint16_t* Yh,
                        int Cin, int Cout, int Tin, int ToutFull, int Nout,
                        int kw, int stride, int pad, int outStride, int outOffset,
                        int eluIn, int eluOut) {
        ConvArgs a;
        a.X = X; a.Wp = Wp; a.bias = bias; a.Yf = Yf; a.Yh = Yh;
        a.Cin = Cin; a.Cout = Cout; a.Tin = Tin; a.ToutFull = ToutFull; a.Nout = Nout;
        a.stride = stride; a.pad = pad;
        a.outStride = outStride; a.outOffset = outOffset;
        a.eluOut = eluOut; a.Ktot = Cin * kw;
        dim3 grid((Nout + 63) / 64, (Cout + 63) / 64, B_SZ);
        if (eluIn) {
            switch (kw) {
            case 1: conv_wmma_kernel<1, true><<<grid, 128, 0, stream>>>(a); break;
            case 2: conv_wmma_kernel<2, true><<<grid, 128, 0, stream>>>(a); break;
            case 3: conv_wmma_kernel<3, true><<<grid, 128, 0, stream>>>(a); break;
            default: conv_wmma_kernel<4, true><<<grid, 128, 0, stream>>>(a); break;
            }
        } else {
            switch (kw) {
            case 1: conv_wmma_kernel<1, false><<<grid, 128, 0, stream>>>(a); break;
            case 2: conv_wmma_kernel<2, false><<<grid, 128, 0, stream>>>(a); break;
            case 3: conv_wmma_kernel<3, false><<<grid, 128, 0, stream>>>(a); break;
            default: conv_wmma_kernel<4, false><<<grid, 128, 0, stream>>>(a); break;
            }
        }
    };
    auto packw = [&](const float* W, uint16_t* Wp, size_t n) {
        cast_f32_bf16_kernel<<<(int)((n + 255) / 256), 256, 0, stream>>>(W, Wp, (int)n);
    };
    auto conv_std = [&](int wIdx, int bIdx, const uint16_t* X, float* Yf, uint16_t* Yh,
                        int Cin, int Cout, int Tin, int kw, int stride, int pad,
                        int eluIn, int eluOut) -> int {
        int Tout = (Tin + 2 * pad - kw) / stride + 1;
        packw(Fp(wIdx), WPK, (size_t)Cout * Cin * kw);
        conv_raw(X, WPK, Fp(bIdx), Yf, Yh, Cin, Cout, Tin, Tout, Tout,
                 kw, stride, pad, 1, 0, eluIn, eluOut);
        return Tout;
    };
    auto conv_tr = [&](int wIdx, int bIdx, const uint16_t* X, float* Yf, uint16_t* Yh,
                       int C, int Tin, int eluOut) -> int {
        int n = C * C;
        pack_convT_kernel<<<(n + 255) / 256, 256, 0, stream>>>(Fp(wIdx), WPK, WPK2, C, C);
        int Tout = 2 * Tin;
        conv_raw(X, WPK,  Fp(bIdx), Yf, Yh, C, C, Tin, Tout, Tin, 2, 1, 1, 2, 0, 0, eluOut);
        conv_raw(X, WPK2, Fp(bIdx), Yf, Yh, C, C, Tin, Tout, Tin, 2, 1, 0, 2, 1, 0, eluOut);
        return Tout;
    };
    // res leaf base: +0 b1, +1 b2, +2 beta, +3 g, +4 w1, +5 w2
    auto resblk = [&](int base, float* xF, uint16_t* xH,
                      float* tmpF, uint16_t* tmpH, float* t2F, int T) {
        conv_std(base + 4, base + 0, xH,  tmpF, tmpH,   CH_SZ, RC_SZ, T, 3, 1, 1, 1, 0);
        conv_std(base + 5, base + 1, tmpH, t2F, nullptr, RC_SZ, CH_SZ, T, 3, 1, 1, 1, 0);
        cwln_kernel<<<B_SZ * T, 128, 0, stream>>>(t2F, xF, Fp(base + 3), Fp(base + 2),
                                                  xF, xH, CH_SZ, T);
    };
    auto encoder = [&](int wIn, int bIn, const int* dB, const int* dW, const int rB[2][2],
                       const uint16_t* inH, int Cin, int Tin,
                       float*& curF, uint16_t*& curH, float*& altF, uint16_t*& altH,
                       float* tmpF, uint16_t* tmpH, float* t2F) -> int {
        int T = conv_std(wIn, bIn, inH, curF, curH, Cin, CH_SZ, Tin, 1, 1, 0, 0, 1);
        for (int d = 0; d < 2; ++d) {
            for (int r = 0; r < 2; ++r) resblk(rB[d][r], curF, curH, tmpF, tmpH, t2F, T);
            conv_std(dW[d], dB[d], curH, altF, altH, CH_SZ, CH_SZ, T, 4, 2, 1, 0, 1);
            T /= 2;
            std::swap(curF, altF); std::swap(curH, altH);
        }
        return T;
    };
    auto decoder = [&](int wIn, int bIn, int wOut, int bOut,
                       const int* uB, const int* uW, const int rB[2][2], int n_res,
                       const uint16_t* inH, int Cin, int Tin, int Cout,
                       float*& curF, uint16_t*& curH, float*& altF, uint16_t*& altH,
                       float* tmpF, uint16_t* tmpH, float* t2F,
                       float* outF, uint16_t* outH) -> int {
        int T = conv_std(wIn, bIn, inH, curF, curH, Cin, CH_SZ, Tin, 1, 1, 0, 0, 1);
        for (int u = 0; u < 2; ++u) {
            T = conv_tr(uW[u], uB[u], curH, altF, altH, CH_SZ, T, 1);
            std::swap(curF, altF); std::swap(curH, altH);
            for (int r = 0; r < n_res; ++r) resblk(rB[u][r], curF, curH, tmpF, tmpH, t2F, T);
        }
        conv_std(wOut, bOut, curH, outF, outH, CH_SZ, Cout, T, 1, 1, 0, 0, 0);
        return T;
    };

    // ---- 0) zero diff accumulators (ws is poisoned; must re-init every call)
    zero_kernel<<<1, 32, 0, stream>>>(diffAcc, 2);

    // ---- 1) embedding -> A5 : h [B,E,L]
    {
        int n = B_SZ * L_SZ;
        embed_kernel<<<(n + 255) / 256, 256, 0, stream>>>(Fp(86), xin, AF[5], AH[5],
                                                          B_SZ, L_SZ, E_SZ);
    }

    // ---- 2) enc0 -> A0 (T=256)
    float* c0F = AF[0]; uint16_t* c0H = AH[0];
    float* a0F = AF[1]; uint16_t* a0H = AH[1];
    {
        const int dB[2] = {88, 102}, dW[2] = {101, 115};
        const int rB[2][2] = {{89, 95}, {103, 109}};
        encoder(116, 87, dB, dW, rB, AH[5], E_SZ, L_SZ,
                c0F, c0H, a0F, a0H, AF[2], AH[2], AF[3]);
    }

    // ---- 3) enc1 -> (T=64)
    float* c1F = AF[1]; uint16_t* c1H = AH[1];
    float* a1F = AF[4]; uint16_t* a1H = AH[4];
    {
        const int dB[2] = {118, 132}, dW[2] = {131, 145};
        const int rB[2][2] = {{119, 125}, {133, 139}};
        encoder(146, 117, dB, dW, rB, c0H, CH_SZ, 256,
                c1F, c1H, a1F, a1H, AF[2], AH[2], AF[3]);
    }

    // ---- 4) qconv1: 512 -> 8, T=64 -> quant_top (raw conv output)
    conv_std(150, 149, c1H, QTf, QTh, CH_SZ, VQ_SZ, 64, 1, 1, 0, 0, 0);

    // ---- 5) VQ top scale -> code_t + diff_t
    const size_t LOGP = (size_t)B_SZ * L_SZ * V_SZ;
    float* code_b_out = out + LOGP + 1;
    float* code_t_out = out + LOGP + 1 + (size_t)B_SZ * (L_SZ / 4);
    {
        int n = B_SZ * 64;
        vq_kernel<<<(n + 255) / 256, 256, 0, stream>>>(
            QTf, Fp(1), code_t_out, diffAcc + 0, B_SZ, VQ_SZ, 64, KCB,
            1.0f / (float)(B_SZ * 64 * VQ_SZ));
    }

    // ---- 6) cond = cond_dec0(quant_top), n_res=1, T=64 -> 256
    {
        float* cF = AF[1]; uint16_t* cH = AH[1];
        float* aF = AF[2]; uint16_t* aH = AH[2];
        const int uB[2] = {4, 12}, uW[2] = {11, 19};
        const int rB[2][2] = {{5, 5}, {13, 13}};
        decoder(20, 2, 21, 3, uB, uW, rB, 1, QTh, VQ_SZ, 64, VQ_SZ,
                cF, cH, aF, aH, AF[3], AH[3], AF[4], CDf, CDh);
    }

    // ---- 7) concat(enc0, cond) -> [B, 520, 256]
    {
        size_t n1 = (size_t)B_SZ * CH_SZ * 256;
        cat_copy_kernel<<<(int)((n1 + 255) / 256), 256, 0, stream>>>(
            c0H, CATh, B_SZ, CH_SZ, 256, CH_SZ + VQ_SZ, 0);
        size_t n2 = (size_t)B_SZ * VQ_SZ * 256;
        cat_copy_kernel<<<(int)((n2 + 255) / 256), 256, 0, stream>>>(
            CDh, CATh, B_SZ, VQ_SZ, 256, CH_SZ + VQ_SZ, CH_SZ);
    }

    // ---- 8) qconv0: 520 -> 8, T=256 -> quant_bottom
    conv_std(148, 147, CATh, QBf, QBh, CH_SZ + VQ_SZ, VQ_SZ, 256, 1, 1, 0, 0, 0);

    // ---- 9) VQ bottom scale -> code_b + diff_b
    {
        int n = B_SZ * 256;
        vq_kernel<<<(n + 255) / 256, 256, 0, stream>>>(
            QBf, Fp(0), code_b_out, diffAcc + 1, B_SZ, VQ_SZ, 256, KCB,
            1.0f / (float)(B_SZ * 256 * VQ_SZ));
    }

    // ---- 10) x1 = dec1(quant_top), T=64 -> 256
    {
        float* cF = AF[1]; uint16_t* cH = AH[1];
        float* aF = AF[2]; uint16_t* aH = AH[2];
        const int uB[2] = {56, 70}, uW[2] = {69, 83};
        const int rB[2][2] = {{57, 63}, {71, 77}};
        decoder(84, 54, 85, 55, uB, uW, rB, 2, QTh, VQ_SZ, 64, VQ_SZ,
                cF, cH, aF, aH, AF[3], AH[3], AF[4], X1f, X1h);
    }

    // ---- 11) concat(quant_bottom, x1) -> [B, 16, 256]
    {
        size_t n = (size_t)B_SZ * VQ_SZ * 256;
        cat_copy_kernel<<<(int)((n + 255) / 256), 256, 0, stream>>>(
            QBh, CAT2h, B_SZ, VQ_SZ, 256, 2 * VQ_SZ, 0);
        cat_copy_kernel<<<(int)((n + 255) / 256), 256, 0, stream>>>(
            X1h, CAT2h, B_SZ, VQ_SZ, 256, 2 * VQ_SZ, VQ_SZ);
    }

    // ---- 12) x0 = dec0(cat2) -> A0 f32 [B, 256, 1024]
    {
        float* cF = AF[1]; uint16_t* cH = AH[1];
        float* aF = AF[2]; uint16_t* aH = AH[2];
        const int uB[2] = {24, 38}, uW[2] = {37, 51};
        const int rB[2][2] = {{25, 31}, {39, 45}};
        decoder(52, 22, 53, 23, uB, uW, rB, 2, CAT2h, 2 * VQ_SZ, 256, V_SZ,
                cF, cH, aF, aH, AF[3], AH[3], AF[4], AF[0], AH[0]);
    }

    // ---- 13) diff = diff_t + diff_b
    diff_finalize_kernel<<<1, 1, 0, stream>>>(diffAcc, out + LOGP);

    // ---- 14) log_softmax over V (TAU = 1.0) -> out[0 .. B*L*V)
    {
        int nw = B_SZ * L_SZ;
        int blocks = (nw * 32 + 127) / 128;
        logsoftmax_kernel<<<blocks, 128, 0, stream>>>(AF[0], out, B_SZ, V_SZ, L_SZ);
    }
}
